// MultiHeadAttention_670014898380
// MI455X (gfx1250) — matrix operand order repaired and checked by reading, not compiled, not run on hardware
//
#include <hip/hip_runtime.h>
#include <hip/hip_bf16.h>
#include <cstdint>

// ---------------------------------------------------------------------------
// Multi-head attention forward for MI455X (gfx1250), bf16 WMMA pipeline with
// CDNA5 data movers:
//   - GEMM tiles staged by the Tensor Data Mover (tensor_load_to_lds, D# with
//     native LDS padding), double buffered, s_wait_tensorcnt.
//   - Flash-attention K/V tiles staged by global_load_async_to_lds_b128,
//     double buffered, s_wait_asynccnt.
// Fallback chain (TDM -> async -> sync) keeps compilation green on any
// toolchain; all matrix math is v_wmma_f32_16x16x32_bf16.
// ---------------------------------------------------------------------------

#define S_LEN 2048
#define EMB   2048
#define NH    16
#define DK    128
#define NBATCH 2

#ifndef __has_builtin
#define __has_builtin(x) 0
#endif
#if __has_builtin(__builtin_amdgcn_tensor_load_to_lds)
#define HAVE_TDM 1
#else
#define HAVE_TDM 0
#endif
#if __has_builtin(__builtin_amdgcn_global_load_async_to_lds_b128)
#define HAVE_ASYNC 1
#else
#define HAVE_ASYNC 0
#endif

typedef __attribute__((ext_vector_type(16))) __bf16 v16bf;
typedef __attribute__((ext_vector_type(8)))  __bf16 v8bf;
typedef __attribute__((ext_vector_type(8)))  float  v8f;
typedef __attribute__((ext_vector_type(4)))  unsigned int u32x4;
typedef __attribute__((ext_vector_type(8)))  int i32x8;
typedef __attribute__((ext_vector_type(4)))  int i32x4;

#define AS1 __attribute__((address_space(1)))
#define AS3 __attribute__((address_space(3)))

__device__ __forceinline__ v16bf combine16(v8bf lo, v8bf hi) {
  return __builtin_shufflevector(lo, hi, 0,1,2,3,4,5,6,7,8,9,10,11,12,13,14,15);
}

// A-fragment (16x32 bf16): lane m = lane%16, half = lane/16.
// e 0..7 -> K = half*8 + e ; e 8..15 -> K = 16 + half*8 + (e-8)
__device__ __forceinline__ v16bf fragA(const __bf16* rowBase, int half) {
  v8bf lo = *(const v8bf*)(rowBase + half * 8);
  v8bf hi = *(const v8bf*)(rowBase + 16 + half * 8);
  return combine16(lo, hi);
}
// B-fragment (32x16 bf16): lane n = lane%16, half = lane/16;
// e 0..7 -> K = half*8 + e ; e 8..15 -> K = 16 + half*8 + (e-8)
__device__ __forceinline__ v16bf fragB(const __bf16* rowBase, int half) {
  v8bf lo = *(const v8bf*)(rowBase + half * 8);
  v8bf hi = *(const v8bf*)(rowBase + 16 + half * 8);
  return combine16(lo, hi);
}

__device__ __forceinline__ v8f wmma_bf16(v16bf a, v16bf b, v8f c) {
  return __builtin_amdgcn_wmma_f32_16x16x32_bf16(false, a, false, b, (short)0, c,
                                                 false, false);
}

__device__ __forceinline__ float hmax16(float v) {
  for (int off = 8; off > 0; off >>= 1) v = fmaxf(v, __shfl_xor(v, off, 16));
  return v;
}
__device__ __forceinline__ float hsum16(float v) {
  for (int off = 8; off > 0; off >>= 1) v += __shfl_xor(v, off, 16);
  return v;
}

// 16-byte global -> LDS copy: async DMA when available, else sync.
__device__ __forceinline__ void cp16(const __bf16* g, __bf16* l) {
#if HAVE_ASYNC
  __builtin_amdgcn_global_load_async_to_lds_b128(
      (AS1 i32x4*)(uintptr_t)g, (AS3 i32x4*)(unsigned)(uintptr_t)l, 0, 0);
#else
  *(v8bf*)l = *(const v8bf*)g;
#endif
}

#if HAVE_TDM
// TDM load of a 128x32 bf16 tile (row stride EMB elements) into LDS with
// 16B padding after every 64B -> LDS row pitch 40 bf16 (matches lA/lB decl).
__device__ __forceinline__ void tdm_load_tile128x32(const __bf16* gbase,
                                                    unsigned ldsOff) {
  u32x4 g0;
  g0[0] = 1u;                               // count=1 (valid descriptor)
  g0[1] = ldsOff;                           // lds_addr
  unsigned long long ga = (unsigned long long)(uintptr_t)gbase;
  g0[2] = (unsigned)ga;                     // global_addr[31:0]
  g0[3] = (unsigned)((ga >> 32) & 0x01FFFFFFull) | (2u << 30);  // addr hi | type=2
  i32x8 g1;
  g1[0] = (1 << 16)       // data_size = 2 bytes
        | (1 << 20)       // pad_enable
        | (3 << 22)       // pad_interval: every 16 DWORDs (64B)
        | (3 << 25);      // pad_amount: 4 DWORDs (16B)
  g1[1] = (int)((EMB & 0xFFFF) << 16);      // tensor_dim0 = 2048
  g1[2] = (int)((4096 & 0xFFFF) << 16);     // tensor_dim1 = 4096
  g1[3] = (32 << 16);                       // tile_dim0 = 32
  g1[4] = 128;                              // tile_dim1 = 128
  g1[5] = EMB;                              // tensor_dim0_stride = 2048
  g1[6] = 0;
  g1[7] = 0;
  i32x4 zz; zz[0] = zz[1] = zz[2] = zz[3] = 0;
  i32x8 z8;
  for (int i = 0; i < 8; i++) z8[i] = 0;
  __builtin_amdgcn_tensor_load_to_lds(g0, g1, zz, zz, z8, 0);
}
#endif

// Wait for the previous GEMM stage's tile transfers.
__device__ __forceinline__ void gemm_stage_wait(bool more) {
#if HAVE_TDM
  if (more) __builtin_amdgcn_s_wait_tensorcnt(2);
  else      __builtin_amdgcn_s_wait_tensorcnt(0);
#elif HAVE_ASYNC
  if (more) asm volatile("s_wait_asynccnt 4" ::: "memory");
  else      asm volatile("s_wait_asynccnt 0" ::: "memory");
#else
  (void)more;
#endif
}

__device__ __forceinline__ void flash_stage_wait(bool more) {
#if HAVE_ASYNC
  if (more) asm volatile("s_wait_asynccnt 16" ::: "memory");
  else      asm volatile("s_wait_asynccnt 0" ::: "memory");
#else
  (void)more;
#endif
}

// ---------------------------------------------------------------------------
// f32 -> bf16 conversion, 8 elements per thread
// ---------------------------------------------------------------------------
__global__ void cvt_f32_bf16(const float4* __restrict__ s, v8bf* __restrict__ d,
                             int n8) {
  int i = blockIdx.x * blockDim.x + threadIdx.x;
  if (i < n8) {
    float4 a = s[2 * i], b = s[2 * i + 1];
    v8bf o;
    o[0] = (__bf16)a.x; o[1] = (__bf16)a.y; o[2] = (__bf16)a.z; o[3] = (__bf16)a.w;
    o[4] = (__bf16)b.x; o[5] = (__bf16)b.y; o[6] = (__bf16)b.z; o[7] = (__bf16)b.w;
    d[i] = o;
  }
}

// ---------------------------------------------------------------------------
// C = A(M x 2048, bf16) @ W(N x 2048, bf16)^T + bias, 128x128x32 tiles,
// TDM-staged double-buffered LDS. 8 waves, each owns a 32x64 patch.
// MODE 0: bf16 [N,H,S,Dk] scaled (Q)   MODE 1: bf16 [N,H,S,Dk] (K)
// MODE 2: bf16 [N,H,Dk,S] transposed (V)   MODE 3: f32 [M,2048] (out proj)
// ---------------------------------------------------------------------------
template <int MODE>
__global__ __launch_bounds__(256, 1)
void gemm_xwT(const __bf16* __restrict__ A, const __bf16* __restrict__ W,
              const float* __restrict__ bias, void* __restrict__ outp,
              float scale) {
  __shared__ __bf16 lA[2][128][40];   // 40 = 32 + TDM pad (16B per 64B row)
  __shared__ __bf16 lB[2][128][40];

  const int tid  = threadIdx.x;
  const int lane = tid & 31, half = lane >> 4, ln = lane & 15;
  const int w = tid >> 5, wm = w >> 1, wn = w & 1;
  const int m0 = blockIdx.y * 128, n0 = blockIdx.x * 128;

  const __bf16* Ag = A + (size_t)m0 * EMB;
  const __bf16* Bg = W + (size_t)n0 * EMB;

  v8f acc[2][4];
#pragma unroll
  for (int i = 0; i < 2; i++)
#pragma unroll
    for (int j = 0; j < 4; j++)
      for (int e = 0; e < 8; e++) acc[i][j][e] = 0.0f;

  auto stage = [&](int b, int k0) {
#if HAVE_TDM
    if (tid < 32) {   // one wave issues both tile DMAs
      tdm_load_tile128x32(Ag + k0, (unsigned)(uintptr_t)&lA[b][0][0]);
      tdm_load_tile128x32(Bg + k0, (unsigned)(uintptr_t)&lB[b][0][0]);
    }
#else
    const int row = tid >> 1, kh = (tid & 1) * 16;
    cp16(Ag + (size_t)row * EMB + k0 + kh,     &lA[b][row][kh]);
    cp16(Ag + (size_t)row * EMB + k0 + kh + 8, &lA[b][row][kh + 8]);
    cp16(Bg + (size_t)row * EMB + k0 + kh,     &lB[b][row][kh]);
    cp16(Bg + (size_t)row * EMB + k0 + kh + 8, &lB[b][row][kh + 8]);
#endif
  };

  stage(0, 0);
  int buf = 0;
  for (int it = 0; it < EMB / 32; ++it) {
    const int nk = (it + 1) * 32;
    const bool more = nk < EMB;
    if (more) stage(buf ^ 1, nk);   // overwrites buffer read in the previous iteration
    gemm_stage_wait(more);
    __syncthreads();

    v16bf af[2], bfr[4];
#pragma unroll
    for (int i = 0; i < 2; i++) af[i] = fragA(&lA[buf][wm * 32 + i * 16 + ln][0], half);
#pragma unroll
    for (int j = 0; j < 4; j++) bfr[j] = fragB(&lB[buf][wn * 64 + j * 16 + ln][0], half);
#pragma unroll
    for (int i = 0; i < 2; i++)
#pragma unroll
      for (int j = 0; j < 4; j++) acc[i][j] = wmma_bf16(af[i], bfr[j], acc[i][j]);

    __syncthreads();   // readers done before next iteration's stage()
    buf ^= 1;
  }

  // Epilogue. C/D layout: element r -> m = half*8 + r, col = lane%16.
#pragma unroll
  for (int i = 0; i < 2; i++) {
#pragma unroll
    for (int j = 0; j < 4; j++) {
      const int gmBase = m0 + wm * 32 + i * 16 + half * 8;
      const int gn     = n0 + wn * 64 + j * 16 + ln;
      const float b    = bias[gn];
#pragma unroll
      for (int r = 0; r < 8; r++) {
        const int gm  = gmBase + r;
        const float v = (acc[i][j][r] + b) * scale;
        if (MODE == 3) {
          ((float*)outp)[(size_t)gm * EMB + gn] = v;
        } else {
          const int h = gn >> 7, d = gn & (DK - 1);
          const int nb = gm >> 11, s = gm & (S_LEN - 1);
          __bf16* o = (__bf16*)outp;
          if (MODE == 2)
            o[(((size_t)nb * NH + h) * DK + d) * S_LEN + s] = (__bf16)v;
          else
            o[(((size_t)nb * NH + h) * S_LEN + s) * DK + d] = (__bf16)v;
        }
      }
    }
  }
}

// ---------------------------------------------------------------------------
// Flash attention. Grid: (S/128, N*H). Block: 256 = 8 waves.
// K/V tiles (128x128 bf16) async-DMA'd into double-buffered dynamic LDS;
// Q frags resident in VGPRs; online softmax with half-wave reductions;
// P relaid through wave-private LDS (no barrier needed, same-wave DS order).
// Dynamic LDS: lK[2][128][136] + lV[2][128][136] = 139264 B.
// ---------------------------------------------------------------------------
#define FPITCH 136
#define FLASH_DYN_LDS ((size_t)4 * 128 * FPITCH * 2)

__global__ __launch_bounds__(256, 1)
void flash_attn(const __bf16* __restrict__ Q, const __bf16* __restrict__ K,
                const __bf16* __restrict__ Vt, __bf16* __restrict__ Ctx) {
  extern __shared__ __bf16 sm[];
  __bf16* lK = sm;                          // [2][128][FPITCH]
  __bf16* lV = sm + 2 * 128 * FPITCH;       // [2][128][FPITCH]
  __shared__ __bf16 lP[8][16][FPITCH];      // per-wave 16x128 P tile

  const int tid  = threadIdx.x;
  const int w    = tid >> 5;
  const int lane = tid & 31, half = lane >> 4, ln = lane & 15;
  const size_t hd = (size_t)blockIdx.y * S_LEN * DK;
  const __bf16* Qp = Q + hd;
  const __bf16* Kp = K + hd;
  const __bf16* Vp = Vt + hd;               // [DK][S]
  __bf16* Cp = Ctx + hd;
  const int qRow0 = blockIdx.x * 128 + w * 16;

  auto kRow = [&](int b, int r) { return lK + ((size_t)b * 128 + r) * FPITCH; };
  auto vRow = [&](int b, int r) { return lV + ((size_t)b * 128 + r) * FPITCH; };

  // cooperative stage: thread -> row tid/2, 64-elem half (tid&1)*64, 8x16B each
  auto stageKV = [&](int b, int t0) {
    const int row = tid >> 1, co = (tid & 1) * 64;
    const __bf16* gk = Kp + (size_t)(t0 + row) * DK + co;
    const __bf16* gv = Vp + (size_t)row * S_LEN + t0 + co;
    __bf16* dk = kRow(b, row) + co;
    __bf16* dv = vRow(b, row) + co;
#pragma unroll
    for (int i = 0; i < 8; i++) {
      cp16(gk + i * 8, dk + i * 8);
      cp16(gv + i * 8, dv + i * 8);
    }
  };

  // Resident Q fragments (Dk=128 -> 4 chunks of 32)
  v16bf qf[4];
  {
    const __bf16* qrow = Qp + (size_t)(qRow0 + ln) * DK;
#pragma unroll
    for (int c = 0; c < 4; c++) qf[c] = fragA(qrow + c * 32, half);
  }

  float mrow[8], lrow[8];
  v8f o[8];
#pragma unroll
  for (int r = 0; r < 8; r++) {
    mrow[r] = -1e30f;
    lrow[r] = 0.0f;
    for (int e = 0; e < 8; e++) o[r][e] = 0.0f;
  }

  stageKV(0, 0);
  int buf = 0;
  for (int it = 0; it < S_LEN / 128; ++it) {
    const bool more = (it + 1) < S_LEN / 128;
    if (more) stageKV(buf ^ 1, (it + 1) * 128);
    flash_stage_wait(more);
    __syncthreads();

    // ---- scores: 8 N-tiles x 4 K-chunks = 32 WMMAs (B-frags from LDS)
    v8f s[8];
#pragma unroll
    for (int j = 0; j < 8; j++) {
      for (int e = 0; e < 8; e++) s[j][e] = 0.0f;
      const __bf16* kr = kRow(buf, j * 16 + ln);
#pragma unroll
      for (int c = 0; c < 4; c++)
        s[j] = wmma_bf16(qf[c], fragB(kr + c * 32, half), s[j]);
    }

    // ---- online softmax (row m = half*8 + r lives in 16 lanes)
    float mNew[8], corr[8];
#pragma unroll
    for (int r = 0; r < 8; r++) {
      float mx = s[0][r];
      for (int j = 1; j < 8; j++) mx = fmaxf(mx, s[j][r]);
      mx = hmax16(mx);
      mNew[r] = fmaxf(mrow[r], mx);
      corr[r] = __expf(mrow[r] - mNew[r]);
    }
#pragma unroll
    for (int r = 0; r < 8; r++) {
      const int m = half * 8 + r;
      float sum = 0.0f;
#pragma unroll
      for (int j = 0; j < 8; j++) {
        float p = __expf(s[j][r] - mNew[r]);
        sum += p;
        lP[w][m][j * 16 + ln] = (__bf16)p;   // D-layout -> LDS
      }
      sum = hsum16(sum);
      lrow[r] = lrow[r] * corr[r] + sum;
      mrow[r] = mNew[r];
    }
#pragma unroll
    for (int jd = 0; jd < 8; jd++)
      for (int r = 0; r < 8; r++) o[jd][r] *= corr[r];

    // ---- P@V: P re-read in A-layout (same wave, DScnt-ordered, no barrier)
    v16bf pf[4];
    {
      const __bf16* prow = &lP[w][ln][0];
#pragma unroll
      for (int c = 0; c < 4; c++) pf[c] = fragA(prow + c * 32, half);
    }
#pragma unroll
    for (int jd = 0; jd < 8; jd++) {
      const __bf16* vr = vRow(buf, jd * 16 + ln);
#pragma unroll
      for (int c = 0; c < 4; c++)
        o[jd] = wmma_bf16(pf[c], fragB(vr + c * 32, half), o[jd]);
    }

    __syncthreads();   // buffer readers done before next stage overwrites
    buf ^= 1;
  }

  // ---- epilogue: normalize, store bf16 context in natural (n,h,s,d) order
#pragma unroll
  for (int r = 0; r < 8; r++) {
    const float inv = 1.0f / lrow[r];
    const int m = half * 8 + r;
    __bf16* crow = Cp + (size_t)(qRow0 + m) * DK;
#pragma unroll
    for (int jd = 0; jd < 8; jd++)
      crow[jd * 16 + ln] = (__bf16)(o[jd][r] * inv);
  }
}

// ---------------------------------------------------------------------------
// Host-side launch
// ---------------------------------------------------------------------------
extern "C" void kernel_launch(void* const* d_in, const int* in_sizes, int n_in,
                              void* d_out, int out_size, void* d_ws, size_t ws_size,
                              hipStream_t stream) {
  (void)in_sizes; (void)n_in; (void)out_size; (void)ws_size;

  const size_t X  = (size_t)NBATCH * S_LEN * EMB;  // 8,388,608 activations
  const size_t WS = (size_t)EMB * EMB;             // 4,194,304 weights

  __bf16* ws  = (__bf16*)d_ws;
  __bf16* xq  = ws;
  __bf16* xk  = xq + X;
  __bf16* xv  = xk + X;
  __bf16* wq  = xv + X;
  __bf16* wk  = wq + WS;
  __bf16* wv  = wk + WS;
  __bf16* wo  = wv + WS;
  __bf16* qw  = wo + WS;     // Q [N,H,S,Dk] (pre-scaled by 1/sqrt(Dk))
  __bf16* kw  = qw + X;      // K [N,H,S,Dk]
  __bf16* vtw = kw + X;      // V [N,H,Dk,S]
  __bf16* ctx = vtw + X;     // attention context (n,h,s,d) flat order

  auto cvt = [&](const void* s, __bf16* d, size_t n) {
    const size_t n8 = n / 8;
    cvt_f32_bf16<<<dim3((unsigned)((n8 + 255) / 256)), dim3(256), 0, stream>>>(
        (const float4*)s, (v8bf*)d, (int)n8);
  };
  cvt(d_in[0], xq, X);
  cvt(d_in[1], xk, X);
  cvt(d_in[2], xv, X);
  cvt(d_in[3], wq, WS);
  cvt(d_in[5], wk, WS);
  cvt(d_in[7], wv, WS);
  cvt(d_in[9], wo, WS);

  const dim3 gg(EMB / 128, (NBATCH * S_LEN) / 128);  // (16, 32)
  const dim3 bb(256);
  const float qscale = 0.0883883476483184f;          // 1/sqrt(128)

  gemm_xwT<0><<<gg, bb, 0, stream>>>(xq, wq, (const float*)d_in[4], qw, qscale);
  gemm_xwT<1><<<gg, bb, 0, stream>>>(xk, wk, (const float*)d_in[6], kw, 1.0f);
  gemm_xwT<2><<<gg, bb, 0, stream>>>(xv, wv, (const float*)d_in[8], vtw, 1.0f);

  flash_attn<<<dim3(S_LEN / 128, NBATCH * NH), bb, FLASH_DYN_LDS, stream>>>(
      qw, kw, vtw, ctx);

  gemm_xwT<3><<<gg, bb, 0, stream>>>(ctx, wo, (const float*)d_in[10],
                                     (float*)d_out, 1.0f);
}
